// DiscreteVAE_62354335203566
// MI455X (gfx1250) — matrix operand order repaired and checked by reading, not compiled, not run on hardware
//
#include <hip/hip_runtime.h>
#include <cmath>

// ---------------------------------------------------------------------------
// CDNA5 WMMA types
// ---------------------------------------------------------------------------
typedef __attribute__((ext_vector_type(16))) _Float16 v16h;
typedef __attribute__((ext_vector_type(8)))  float    v8f;

#define TM 128
#define TN 64
#define TK 32
#define LDSP (TK + 8)

// ---------------------------------------------------------------------------
// Implicit-GEMM convolution via v_wmma_f32_16x16x32_f16, templated on the
// (square) spatial kernel size KS so all k-decodes are divisions by
// compile-time constants (1x1: k == ci, no decode at all).
// GEMM view: M = B*Hout*Wout, N = Cout, K = Cin*KS*KS.
// Block: 256 threads = 8 waves (4 M-slices x 2 N-slices), 128x64 output tile.
// Per wave: 32x32 subtile = 4 accumulators, 4 WMMAs per K-step.
// Double-buffered LDS with register-staged global prefetch of the next tile,
// plus global_prefetch of the weight stream two tiles ahead.
// Optional fused residual add (skip) in the epilogue.
// ---------------------------------------------------------------------------
template <int KS>
__global__ __launch_bounds__(256) void conv_wmma_kernel(
    const float* __restrict__ in, const float* __restrict__ wgt,
    const float* __restrict__ bias, const float* __restrict__ skip,
    float* __restrict__ out,
    int Bn, int Cin, int Hin, int Win, int Cout, int Hout, int Wout,
    int stride, int padT, int padL)
{
  const int M = Bn * Hout * Wout;
  const int N = Cout;
  const int K = Cin * KS * KS;
  const int m0 = blockIdx.x * TM;
  const int n0 = blockIdx.y * TN;

  __shared__ _Float16 As[2][TM][LDSP];   // [buf][row][k]
  __shared__ _Float16 Bs[2][TN][LDSP];   // [buf][col][k]

  const int tid  = threadIdx.x;
  const int lane = tid & 31;
  const int wave = tid >> 5;
  const int wm   = wave & 3;   // M subtile 0..3 (32 rows each)
  const int wn   = wave >> 2;  // N subtile 0..1 (32 cols each)

  // ---- A staging: thread owns fixed row (coalesced along ow), walks K ----
  const int ar   = tid & 127;        // row within tile
  const int akk0 = tid >> 7;         // starting k (0/1), step 2 -> 16 elems
  int a_b = 0, a_oh = 0, a_ow = 0;
  bool a_mok = false;
  {
    const int m = m0 + ar;
    a_mok = m < M;
    if (a_mok) {
      a_ow = m % Wout;
      const int t = m / Wout;
      a_oh = t % Hout;
      a_b  = t / Hout;
    }
  }
  const int a_ih0 = a_oh * stride - padT;   // + kh
  const int a_iw0 = a_ow * stride - padL;   // + kw
  // ---- B staging: consecutive tids -> consecutive k (coalesced), 8 elems ----
  const int bkk = tid & 31;
  const int bn0 = tid >> 5;

  float ra[16];
  float rb[8];

  auto loadA = [&](int k0) {
    #pragma unroll
    for (int j = 0; j < 16; ++j) {
      const int kk = akk0 + 2 * j;
      const int k  = k0 + kk;
      float v = 0.0f;
      if (a_mok && k < K) {
        if (KS == 1) {
          // 1x1: k == ci, spatial position is (a_ih0, a_iw0) itself
          if (a_ih0 >= 0 && a_ih0 < Hin && a_iw0 >= 0 && a_iw0 < Win)
            v = in[(((long)a_b * Cin + k) * Hin + a_ih0) * Win + a_iw0];
        } else {
          const int kw = k % KS;          // divisions by constant -> mul/shift
          const int t2 = k / KS;
          const int kh = t2 % KS;
          const int ci = t2 / KS;
          const int ih = a_ih0 + kh;
          const int iw = a_iw0 + kw;
          if (ih >= 0 && ih < Hin && iw >= 0 && iw < Win)
            v = in[(((long)a_b * Cin + ci) * Hin + ih) * Win + iw];
        }
      }
      ra[j] = v;
    }
  };
  auto loadB = [&](int k0) {
    #pragma unroll
    for (int j = 0; j < 8; ++j) {
      const int n = n0 + bn0 + 8 * j;
      const int k = k0 + bkk;
      float v = 0.0f;
      if (n < N && k < K) v = wgt[(long)n * K + k];
      rb[j] = v;
    }
    // pull the weight stream two K-tiles ahead into cache (global_prefetch)
    const int kpf = k0 + TK + bkk;
    if (kpf < K && (n0 + bn0) < N)
      __builtin_prefetch(&wgt[(long)(n0 + bn0) * K + kpf], 0, 1);
  };
  auto storeA = [&](int buf) {
    #pragma unroll
    for (int j = 0; j < 16; ++j)
      As[buf][ar][akk0 + 2 * j] = (_Float16)ra[j];
  };
  auto storeB = [&](int buf) {
    #pragma unroll
    for (int j = 0; j < 8; ++j)
      Bs[buf][bn0 + 8 * j][bkk] = (_Float16)rb[j];
  };

  v8f acc00 = {}, acc01 = {}, acc10 = {}, acc11 = {};

  loadA(0);
  loadB(0);
  storeA(0);
  storeB(0);
  __syncthreads();

  int cbuf = 0;
  for (int k0 = 0; k0 < K; k0 += TK) {
    const bool more = (k0 + TK) < K;
    // prefetch next tile into registers (overlaps with WMMAs below)
    if (more) { loadA(k0 + TK); loadB(k0 + TK); }

    // ---- fragments from current LDS buffer ----
    // A layout: lane<16 holds K{0..7,16..23}, lane>=16 holds K{8..15,24..31}
    const int arow0 = wm * 32 + (lane & 15);
    const int akb   = (lane < 16) ? 0 : 8;
    v16h a0, a1;
    #pragma unroll
    for (int i = 0; i < 8; ++i) {
      a0[i]     = As[cbuf][arow0][akb + i];
      a0[i + 8] = As[cbuf][arow0][akb + 16 + i];
      a1[i]     = As[cbuf][arow0 + 16][akb + i];
      a1[i + 8] = As[cbuf][arow0 + 16][akb + 16 + i];
    }
    // B layout: lane<16 -> col=lane, K{0..7,16..23} ; lane>=16 -> col, K{8..15,24..31}
    const int bcol = wn * 32 + (lane & 15);
    const int bks  = (lane < 16) ? 0 : 8;
    v16h b0, b1;
    #pragma unroll
    for (int i = 0; i < 16; ++i) {
      b0[i] = Bs[cbuf][bcol][bks + i + (i >= 8 ? 8 : 0)];
      b1[i] = Bs[cbuf][bcol + 16][bks + i + (i >= 8 ? 8 : 0)];
    }
    acc00 = __builtin_amdgcn_wmma_f32_16x16x32_f16(false, a0, false, b0,
                                                   (short)0, acc00, false, false);
    acc01 = __builtin_amdgcn_wmma_f32_16x16x32_f16(false, a0, false, b1,
                                                   (short)0, acc01, false, false);
    acc10 = __builtin_amdgcn_wmma_f32_16x16x32_f16(false, a1, false, b0,
                                                   (short)0, acc10, false, false);
    acc11 = __builtin_amdgcn_wmma_f32_16x16x32_f16(false, a1, false, b1,
                                                   (short)0, acc11, false, false);

    if (more) { storeA(cbuf ^ 1); storeB(cbuf ^ 1); }
    __syncthreads();
    cbuf ^= 1;
  }

  // ---- epilogue: C/D layout VGPR r, lanes 0-15 -> M=r,N=lane; 16-31 -> M=r+8.
  // Decode (b,oh,ow) once per fragment, then carry-increment across rows. ----
  const int nc0 = n0 + wn * 32 + (lane & 15);
  const int nc1 = nc0 + 16;
  const float bia0 = (nc0 < N) ? bias[nc0] : 0.0f;
  const float bia1 = (nc1 < N) ? bias[nc1] : 0.0f;
  #pragma unroll
  for (int mi = 0; mi < 2; ++mi) {
    const int mbase = m0 + wm * 32 + mi * 16 + ((lane >= 16) ? 8 : 0);
    const v8f vc0 = mi ? acc10 : acc00;
    const v8f vc1 = mi ? acc11 : acc01;
    int ow = 0, oh = 0, bb = 0;
    if (mbase < M) {
      ow = mbase % Wout;
      const int t = mbase / Wout;
      oh = t % Hout;
      bb = t / Hout;
    }
    #pragma unroll
    for (int r = 0; r < 8; ++r) {
      if (mbase + r < M) {
        const long sp = (long)oh * Wout + ow;
        if (nc0 < N) {
          const long idx = ((long)bb * N + nc0) * ((long)Hout * Wout) + sp;
          float v = vc0[r] + bia0;
          if (skip) v += skip[idx];
          out[idx] = v;
        }
        if (nc1 < N) {
          const long idx = ((long)bb * N + nc1) * ((long)Hout * Wout) + sp;
          float v = vc1[r] + bia1;
          if (skip) v += skip[idx];
          out[idx] = v;
        }
      }
      // carry-increment ow -> oh -> b
      if (++ow == Wout) { ow = 0; if (++oh == Hout) { oh = 0; ++bb; } }
    }
  }
}

// ---------------------------------------------------------------------------
// GroupNorm + fused activation. grid = (B, groups). act: 0 none, 1 swish,
// 2 relu, 3 leaky(0.2). In-place safe.
// ---------------------------------------------------------------------------
__global__ __launch_bounds__(256) void groupnorm_act_kernel(
    const float* __restrict__ in, float* __restrict__ out,
    const float* __restrict__ gamma, const float* __restrict__ beta,
    int C, int HW, int groups, int act)
{
  const int b   = blockIdx.x;
  const int g   = blockIdx.y;
  const int tid = threadIdx.x;
  const int cpg = C / groups;
  const long n  = (long)cpg * HW;
  const float* base  = in  + ((long)b * C + (long)g * cpg) * HW;
  float*       obase = out + ((long)b * C + (long)g * cpg) * HW;

  float s = 0.0f, ss = 0.0f;
  for (long i = tid; i < n; i += 256) {
    const float v = base[i];
    s += v; ss += v * v;
  }
  __shared__ float r0[256], r1[256];
  r0[tid] = s; r1[tid] = ss;
  __syncthreads();
  for (int off = 128; off > 0; off >>= 1) {
    if (tid < off) { r0[tid] += r0[tid + off]; r1[tid] += r1[tid + off]; }
    __syncthreads();
  }
  const float mean = r0[0] / (float)n;
  const float var  = r1[0] / (float)n - mean * mean;
  const float rstd = rsqrtf(var + 1e-5f);

  for (long i = tid; i < n; i += 256) {
    const int c = g * cpg + (int)(i / HW);
    float v = (base[i] - mean) * rstd * gamma[c] + beta[c];
    if (act == 1)      v = v / (1.0f + __expf(-v));      // swish
    else if (act == 2) v = fmaxf(v, 0.0f);               // relu
    else if (act == 3) v = (v > 0.0f) ? v : 0.2f * v;    // leaky 0.2
    obase[i] = v;
  }
}

// ---------------------------------------------------------------------------
// Attention scores + row softmax. grid = (N, B), block = 256 (= N keys).
// q,k layout (B, C, N). w layout (B, N, N).
// ---------------------------------------------------------------------------
__global__ __launch_bounds__(256) void attn_scores_softmax_kernel(
    const float* __restrict__ q, const float* __restrict__ k,
    float* __restrict__ w, int C, int Nn, float scale)
{
  const int b   = blockIdx.y;
  const int qi  = blockIdx.x;
  const int tid = threadIdx.x;
  __shared__ float qcol[512];
  for (int c = tid; c < C; c += 256)
    qcol[c] = q[((long)b * C + c) * Nn + qi];
  __syncthreads();

  float s = 0.0f;
  for (int c = 0; c < C; ++c)
    s += qcol[c] * k[((long)b * C + c) * Nn + tid];
  s *= scale;

  __shared__ float red[256];
  red[tid] = s;
  __syncthreads();
  for (int off = 128; off > 0; off >>= 1) {
    if (tid < off) red[tid] = fmaxf(red[tid], red[tid + off]);
    __syncthreads();
  }
  const float mx = red[0];
  __syncthreads();
  const float e = __expf(s - mx);
  red[tid] = e;
  __syncthreads();
  for (int off = 128; off > 0; off >>= 1) {
    if (tid < off) red[tid] += red[tid + off];
    __syncthreads();
  }
  w[((long)b * Nn + qi) * Nn + tid] = e / red[0];
}

// o[b,d,q] = sum_k w[b,q,k] * v[b,d,k]. grid = (C, B), block = 256 (= N).
__global__ __launch_bounds__(256) void attn_out_kernel(
    const float* __restrict__ w, const float* __restrict__ v,
    float* __restrict__ o, int C, int Nn)
{
  const int b   = blockIdx.y;
  const int d   = blockIdx.x;
  const int tid = threadIdx.x;
  __shared__ float vrow[256];
  vrow[tid] = v[((long)b * C + d) * Nn + tid];
  __syncthreads();
  float acc = 0.0f;
  const float* wrow = w + ((long)b * Nn + tid) * Nn;
  for (int kk = 0; kk < Nn; ++kk) acc += wrow[kk] * vrow[kk];
  o[((long)b * C + d) * Nn + tid] = acc;
}

// ---------------------------------------------------------------------------
// Nearest-codebook quantize (STE forward == codebook row). C == 256 == block.
// grid = B*HW blocks.
// ---------------------------------------------------------------------------
__global__ __launch_bounds__(256) void quantize_kernel(
    const float* __restrict__ z, const float* __restrict__ cb,
    float* __restrict__ zq, int C, int HW, int nE)
{
  const int p   = blockIdx.x;
  const int b   = p / HW;
  const int hw  = p % HW;
  const int tid = threadIdx.x;
  __shared__ float zf[256];
  zf[tid] = z[((long)b * C + tid) * HW + hw];
  __syncthreads();

  float best = 3.4e38f;
  int   bi   = 0x7fffffff;
  for (int e = tid; e < nE; e += 256) {
    const float* cr = cb + (long)e * C;
    float d = 0.0f;
    for (int c = 0; c < C; ++c) { const float t = zf[c] - cr[c]; d += t * t; }
    if (d < best) { best = d; bi = e; }
  }
  __shared__ float rd[256];
  __shared__ int   ri[256];
  rd[tid] = best; ri[tid] = bi;
  __syncthreads();
  for (int off = 128; off > 0; off >>= 1) {
    if (tid < off) {
      if (rd[tid + off] < rd[tid] ||
          (rd[tid + off] == rd[tid] && ri[tid + off] < ri[tid])) {
        rd[tid] = rd[tid + off];
        ri[tid] = ri[tid + off];
      }
    }
    __syncthreads();
  }
  const int idx = ri[0];
  zq[((long)b * C + tid) * HW + hw] = cb[(long)idx * C + tid];
}

// ===========================================================================
// Host-side parameter walking (JAX pytree leaf order: dict keys sorted).
// ===========================================================================
struct ConvP { const float* b; const float* w; };
struct NormP { const float* b; const float* g; };
struct ResP  { ConvP c1, c2, nin; NormP n1, n2; bool has_nin; };
struct AttnP { ConvP k; NormP norm; ConvP proj; ConvP q; ConvP v; };

struct Cur {
  void* const* a; int i;
  const float* next() { return (const float*)a[i++]; }
};

static ConvP rdConv(Cur& c) { ConvP p; p.b = c.next(); p.w = c.next(); return p; }
static NormP rdNorm(Cur& c) { NormP p; p.b = c.next(); p.g = c.next(); return p; }
static ResP  rdRes(Cur& c, bool nin) {
  ResP r; r.c1 = rdConv(c); r.c2 = rdConv(c);
  r.has_nin = nin; if (nin) r.nin = rdConv(c);
  r.n1 = rdNorm(c); r.n2 = rdNorm(c); return r;
}
static AttnP rdAttn(Cur& c) {
  AttnP a; a.k = rdConv(c); a.norm = rdNorm(c); a.proj = rdConv(c);
  a.q = rdConv(c); a.v = rdConv(c); return a;
}

static void conv_l(hipStream_t s, const float* in, const ConvP& p, float* out,
                   int Bn, int Cin, int Hin, int Win, int Cout,
                   int KHW, int stride, int padT, int padL,
                   int Hout, int Wout, const float* skip = nullptr)
{
  const int M = Bn * Hout * Wout;
  dim3 grid((unsigned)((M + TM - 1) / TM), (unsigned)((Cout + TN - 1) / TN));
  if (KHW == 3)
    conv_wmma_kernel<3><<<grid, 256, 0, s>>>(in, p.w, p.b, skip, out, Bn, Cin,
                                             Hin, Win, Cout, Hout, Wout,
                                             stride, padT, padL);
  else
    conv_wmma_kernel<1><<<grid, 256, 0, s>>>(in, p.w, p.b, skip, out, Bn, Cin,
                                             Hin, Win, Cout, Hout, Wout,
                                             stride, padT, padL);
}

static void gn_l(hipStream_t s, const float* in, float* out, const NormP& p,
                 int Bn, int C, int HW, int groups, int act)
{
  dim3 grid((unsigned)Bn, (unsigned)groups);
  groupnorm_act_kernel<<<grid, 256, 0, s>>>(in, out, p.g, p.b, C, HW, groups, act);
}

static void resblock_l(hipStream_t s, float* bufs[3], int& hi, float* ninbuf,
                       const ResP& p, int Bn, int Cin, int Cout, int H, int W)
{
  int j = -1, k = -1;
  for (int i = 0; i < 3; ++i) if (i != hi) { if (j < 0) j = i; else k = i; }
  float* x  = bufs[hi];
  float* t1 = bufs[j];
  float* t2 = bufs[k];
  const int HW = H * W;
  gn_l(s, x, t1, p.n1, Bn, Cin, HW, 32, 1);
  conv_l(s, t1, p.c1, t2, Bn, Cin, H, W, Cout, 3, 1, 1, 1, H, W);
  gn_l(s, t2, t2, p.n2, Bn, Cout, HW, 32, 1);
  const float* skip = x;
  if (p.has_nin) {
    conv_l(s, x, p.nin, ninbuf, Bn, Cin, H, W, Cout, 1, 1, 0, 0, H, W);
    skip = ninbuf;
  }
  // conv2 with fused residual add -> t1
  conv_l(s, t2, p.c2, t1, Bn, Cout, H, W, Cout, 3, 1, 1, 1, H, W, skip);
  hi = j;
}

static void attnblock_l(hipStream_t s, float* bufs[3], int& hi,
                        float* qb, float* kb, float* vb, float* wb,
                        const AttnP& p, int Bn, int C, int H, int W)
{
  int j = -1, k = -1;
  for (int i = 0; i < 3; ++i) if (i != hi) { if (j < 0) j = i; else k = i; }
  float* x  = bufs[hi];
  float* t1 = bufs[j];
  float* t2 = bufs[k];
  const int Nn = H * W;
  gn_l(s, x, t1, p.norm, Bn, C, Nn, 32, 0);
  conv_l(s, t1, p.q, qb, Bn, C, H, W, C, 1, 1, 0, 0, H, W);
  conv_l(s, t1, p.k, kb, Bn, C, H, W, C, 1, 1, 0, 0, H, W);
  conv_l(s, t1, p.v, vb, Bn, C, H, W, C, 1, 1, 0, 0, H, W);
  dim3 g1((unsigned)Nn, (unsigned)Bn);
  attn_scores_softmax_kernel<<<g1, 256, 0, s>>>(qb, kb, wb, C, Nn,
                                                1.0f / sqrtf((float)C));
  dim3 g2((unsigned)C, (unsigned)Bn);
  attn_out_kernel<<<g2, 256, 0, s>>>(wb, vb, t1, C, Nn);
  // proj with fused residual add -> t2
  conv_l(s, t1, p.proj, t2, Bn, C, H, W, C, 1, 1, 0, 0, H, W, x);
  hi = k;
}

// ===========================================================================
extern "C" void kernel_launch(void* const* d_in, const int* in_sizes, int n_in,
                              void* d_out, int out_size, void* d_ws, size_t ws_size,
                              hipStream_t stream)
{
  (void)in_sizes; (void)n_in; (void)out_size; (void)ws_size;
  Cur cur{d_in, 0};

  // ---- inputs: x first (setup_inputs insertion order), then params tree ----
  const float* x        = cur.next();           // (2, 3, 256, 256)
  const float* codebook = cur.next();           // params: 'codebook' < 'dec' < 'enc'

  // ---- dec (sorted: conv_in, conv_out, mid1, mid2, mid_attn, norm_out, ups) ----
  ConvP dec_conv_in  = rdConv(cur);
  ConvP dec_conv_out = rdConv(cur);
  ResP  dec_mid1     = rdRes(cur, false);
  ResP  dec_mid2     = rdRes(cur, false);
  AttnP dec_mid_attn = rdAttn(cur);
  NormP dec_norm_out = rdNorm(cur);

  struct DecLvl { bool attn; ResP r[3]; AttnP a[3]; bool has_up; ConvP up; };
  DecLvl ups[5];
  const bool u_attn[5] = { true,  false, false, false, false };
  const bool u_nin0[5] = { false, true,  false, true,  false };
  const bool u_up[5]   = { true,  true,  true,  true,  false };
  for (int L = 0; L < 5; ++L) {
    for (int bk = 0; bk < 3; ++bk) {
      if (u_attn[L]) ups[L].a[bk] = rdAttn(cur);          // 'attn' < 'res'
      ups[L].r[bk] = rdRes(cur, bk == 0 && u_nin0[L]);
    }
    ups[L].attn   = u_attn[L];
    ups[L].has_up = u_up[L];
    if (u_up[L]) ups[L].up = rdConv(cur);                 // 'blocks' < 'up'
  }

  // ---- enc (sorted: conv_in, conv_out, levels, mid1, mid2, mid_attn, norm_out) ----
  ConvP enc_conv_in  = rdConv(cur);
  ConvP enc_conv_out = rdConv(cur);
  struct EncLvl { bool attn; ResP r[2]; AttnP a[2]; bool has_down; ConvP down; };
  EncLvl levels[5];
  const bool e_attn[5] = { false, false, false, false, true  };
  const bool e_nin0[5] = { false, false, true,  false, true  };
  const bool e_down[5] = { true,  true,  true,  true,  false };
  for (int L = 0; L < 5; ++L) {
    for (int bk = 0; bk < 2; ++bk) {
      if (e_attn[L]) levels[L].a[bk] = rdAttn(cur);
      levels[L].r[bk] = rdRes(cur, bk == 0 && e_nin0[L]);
    }
    levels[L].attn     = e_attn[L];
    levels[L].has_down = e_down[L];
    if (e_down[L]) levels[L].down = rdConv(cur);
  }
  ResP  enc_mid1     = rdRes(cur, false);
  ResP  enc_mid2     = rdRes(cur, false);
  AttnP enc_mid_attn = rdAttn(cur);
  NormP enc_norm_out = rdNorm(cur);

  // ---- workspace layout (floats) ----
  const size_t BIGE = 16777216ull;   // 2*128*256*256
  float* ws = (float*)d_ws;
  float* bufs[3] = { ws, ws + BIGE, ws + 2 * BIGE };
  float* ninb = ws + 3 * BIGE;                   // <= 2*256*64*64
  float* qb   = ninb + 2097152;                  // 2*512*256 each
  float* kb   = qb + 262144;
  float* vb   = kb + 262144;
  float* wb   = vb + 262144;                     // 2*256*256

  const int Bn = 2;
  hipStream_t s = stream;

  // ============================ ENCODER ============================
  int Hc = 256, Wc = 256, Cc = 128;
  conv_l(s, x, enc_conv_in, bufs[0], Bn, 3, 256, 256, 128, 3, 1, 1, 1, 256, 256);
  int hi = 0;
  const int e_out[5] = { 128, 128, 256, 256, 512 };
  for (int L = 0; L < 5; ++L) {
    for (int bk = 0; bk < 2; ++bk) {
      const int cin = Cc, cout = e_out[L];
      resblock_l(s, bufs, hi, ninb, levels[L].r[bk], Bn, cin, cout, Hc, Wc);
      Cc = cout;
      if (levels[L].attn)
        attnblock_l(s, bufs, hi, qb, kb, vb, wb, levels[L].a[bk], Bn, Cc, Hc, Wc);
    }
    if (levels[L].has_down) {
      const int j = (hi + 1) % 3;
      // Conv2dSame k=3 s=2, pad (top,left)=(0,0), (bottom,right)=(1,1)
      conv_l(s, bufs[hi], levels[L].down, bufs[j], Bn, Cc, Hc, Wc, Cc,
             3, 2, 0, 0, Hc / 2, Wc / 2);
      hi = j; Hc /= 2; Wc /= 2;
    }
  }
  resblock_l(s, bufs, hi, ninb, enc_mid1, Bn, Cc, Cc, Hc, Wc);
  attnblock_l(s, bufs, hi, qb, kb, vb, wb, enc_mid_attn, Bn, Cc, Hc, Wc);
  resblock_l(s, bufs, hi, ninb, enc_mid2, Bn, Cc, Cc, Hc, Wc);
  {
    const int j = (hi + 1) % 3;
    gn_l(s, bufs[hi], bufs[j], enc_norm_out, Bn, Cc, Hc * Wc, Cc, 2); // groups=C, relu
    const int k2 = (hi + 2) % 3;
    conv_l(s, bufs[j], enc_conv_out, bufs[k2], Bn, Cc, Hc, Wc, 256,
           3, 1, 1, 1, Hc, Wc);
    hi = k2; Cc = 256;
  }

  // ============================ QUANTIZE ============================
  {
    const int j = (hi + 1) % 3;
    quantize_kernel<<<(unsigned)(Bn * Hc * Wc), 256, 0, s>>>(
        bufs[hi], codebook, bufs[j], 256, Hc * Wc, 1024);
    hi = j;
  }

  // ============================ DECODER ============================
  {
    const int j = (hi + 1) % 3;
    conv_l(s, bufs[hi], dec_conv_in, bufs[j], Bn, 256, Hc, Wc, 512,
           3, 1, 1, 1, Hc, Wc);
    hi = j; Cc = 512;
  }
  resblock_l(s, bufs, hi, ninb, dec_mid1, Bn, Cc, Cc, Hc, Wc);
  attnblock_l(s, bufs, hi, qb, kb, vb, wb, dec_mid_attn, Bn, Cc, Hc, Wc);
  resblock_l(s, bufs, hi, ninb, dec_mid2, Bn, Cc, Cc, Hc, Wc);

  const int d_outc[5] = { 512, 256, 256, 128, 128 };
  for (int L = 0; L < 5; ++L) {
    for (int bk = 0; bk < 3; ++bk) {
      const int cin = Cc, cout = d_outc[L];
      resblock_l(s, bufs, hi, ninb, ups[L].r[bk], Bn, cin, cout, Hc, Wc);
      Cc = cout;
      if (ups[L].attn)
        attnblock_l(s, bufs, hi, qb, kb, vb, wb, ups[L].a[bk], Bn, Cc, Hc, Wc);
    }
    if (ups[L].has_up) {
      // Upsample is conv-only (no spatial resize) per reference
      const int j = (hi + 1) % 3;
      conv_l(s, bufs[hi], ups[L].up, bufs[j], Bn, Cc, Hc, Wc, Cc,
             3, 1, 1, 1, Hc, Wc);
      hi = j;
    }
  }
  {
    const int j = (hi + 1) % 3;
    gn_l(s, bufs[hi], bufs[j], dec_norm_out, Bn, Cc, Hc * Wc, Cc, 3); // groups=C, leaky
    conv_l(s, bufs[j], dec_conv_out, (float*)d_out, Bn, Cc, Hc, Wc, 3,
           3, 1, 1, 1, Hc, Wc);
  }
}
